// BaseRVBackbone_25778393711234
// MI455X (gfx1250) — matrix operand order repaired and checked by reading, not compiled, not run on hardware
//
#include <hip/hip_runtime.h>
#include <hip/hip_bf16.h>

typedef __attribute__((ext_vector_type(16))) _Float16 v16h;
typedef __attribute__((ext_vector_type(8)))  _Float16 v8h;
typedef __attribute__((ext_vector_type(8)))  float    v8f;

#define PI_F 3.14159f
#define FOVU (3.0f * PI_F / 180.0f)
#define FOVD (25.0f * PI_F / 180.0f)
#define BB 8
#define HH 48
#define WW 2048
#define CC 64
#define FW 1024
#define FC0 512
#define NPTS (8 * 102400)

union V16U { v16h v; v8h h2[2]; };

// Async global->LDS 16-byte copy (CDNA5, ASYNCcnt-tracked).
__device__ __forceinline__ void async_copy_b128(void* lds_dst, const void* gsrc) {
    uint32_t laddr = (uint32_t)(uintptr_t)lds_dst;   // LDS offset = addr[31:0]
    uint64_t gaddr = (uint64_t)(uintptr_t)gsrc;
    asm volatile("global_load_async_to_lds_b128 %0, %1, off"
                 :: "v"(laddr), "v"(gaddr) : "memory");
}
__device__ __forceinline__ void wait_asynccnt0() {
    asm volatile("s_wait_asynccnt 0x0" ::: "memory");
}

// ---------------------------------------------------------------- utilities
__global__ void zero_kernel(float4* __restrict__ p, long n4) {
    long i = (long)blockIdx.x * blockDim.x + threadIdx.x;
    long stride = (long)gridDim.x * blockDim.x;
    float4 z = make_float4(0.f, 0.f, 0.f, 0.f);
    for (; i < n4; i += stride) p[i] = z;
}

__device__ __forceinline__ void project_pt(const float* __restrict__ cp, int i,
                                           int& bi, int& u, int& v) {
    float b = cp[(size_t)i * 4 + 0];
    float x = cp[(size_t)i * 4 + 1];
    float y = cp[(size_t)i * 4 + 2];
    float z = cp[(size_t)i * 4 + 3];
    float r = sqrtf(x * x + y * y + z * z);
    float uf = 0.5f * (1.0f - atan2f(y, x) / PI_F) * (float)WW;
    float vf = (1.0f - (asinf(z / r) + FOVD) / (FOVU + FOVD)) * (float)HH;
    uf = fminf(fmaxf(uf, 0.0f), (float)(WW - 1));
    vf = fminf(fmaxf(vf, 0.0f), (float)(HH - 1));
    bi = (int)b;
    u = (int)uf;
    v = (int)vf;
}

// ------------------------------------------------------- scatter points -> F
__global__ void scatter_kernel(const float* __restrict__ cp,
                               const float* __restrict__ pf,
                               float* __restrict__ F) {
    int i = blockIdx.x * blockDim.x + threadIdx.x;
    if (i >= NPTS) return;
    int bi, u, v;
    project_pt(cp, i, bi, u, v);
    if (u >= FC0 && u < FC0 + FW) {
        size_t dst = ((((size_t)bi * HH + v) * FW) + (u - FC0)) * CC;
        const float4* s = (const float4*)(pf + (size_t)i * CC);
        float4* d = (float4*)(F + dst);
#pragma unroll
        for (int c = 0; c < 16; ++c) d[c] = s[c];
    }
}

// ------------------------------------------- dilated 3x3 conv, implicit GEMM
// Input: fp32 (IN_F32=true, conv1 from F) or f16 (conv2/conv3 from x buffers).
// Output: f16 NHWC.  w: (3,3,64,64) HWIO fp32.
// WG = 256 thr = 8 waves; tile = 32 px x 64 oc; wave tile = 16 px x 16 oc.
template <bool IN_F32>
__global__ __launch_bounds__(256) void conv3x3_kernel(const void* __restrict__ inp,
                                                      const float* __restrict__ w,
                                                      _Float16* __restrict__ out,
                                                      int d) {
    __shared__ _Float16 wlds[9 * 64 * 64];     // [tap][co][ci], 72 KB
    __shared__ _Float16 in_lds[3 * 38 * 64];   // [ky][col][ci], 14.25 KB
    __shared__ _Float16 out_lds[32 * 64];      // [px][oc], 4 KB store-repack

    const int tid   = threadIdx.x;
    const int b     = blockIdx.z;
    const int wbase = blockIdx.x * 32;
    const int h0    = blockIdx.y * 12;

    // weights fp32 -> f16, transposed so B fragments are contiguous in ci
    for (int t = tid; t < 9 * 64 * 64; t += 256) {
        int tap = t >> 12;
        int rem = t & 4095;
        int co  = rem >> 6;
        int ci  = rem & 63;
        wlds[t] = (_Float16)w[((size_t)tap * 64 + ci) * 64 + co];
    }

    const int lane = tid & 31, wave = tid >> 5;
    const int m = lane & 15, g = lane >> 4;
    const int pxg = wave & 1, ocg = wave >> 1;
    const int width = 32 + 2 * d;

    for (int h = h0; h < h0 + 12; ++h) {
        __syncthreads();
        // stage 3 input rows (f16), zero-padded at crop edges
        if (IN_F32) {
            const float* in = (const float*)inp;
            const int nstage = 3 * width * 64;
            for (int t = tid; t < nstage; t += 256) {
                int ky  = t / (width * 64);
                int rem = t - ky * width * 64;
                int col = rem >> 6;
                int ci  = rem & 63;
                int gr  = h + (ky - 1) * d;
                int gc  = wbase - d + col;
                float val = 0.0f;
                if (gr >= 0 && gr < HH && gc >= 0 && gc < FW)
                    val = in[(((size_t)b * HH + gr) * FW + gc) * 64 + ci];
                in_lds[(ky * 38 + col) * 64 + ci] = (_Float16)val;
            }
        } else {
            // f16 input: async global->LDS copies; zeros via ds_store for halo
            const _Float16* in = (const _Float16*)inp;
            const int nvec = 3 * width * 8;  // 8 halves per copy
            for (int t = tid; t < nvec; t += 256) {
                int idx = t * 8;
                int ky  = idx / (width * 64);
                int rem = idx - ky * width * 64;
                int col = rem >> 6;
                int ci  = rem & 63;
                int gr  = h + (ky - 1) * d;
                int gc  = wbase - d + col;
                _Float16* ldst = &in_lds[(ky * 38 + col) * 64 + ci];
                if (gr >= 0 && gr < HH && gc >= 0 && gc < FW) {
                    async_copy_b128(ldst,
                        &in[(((size_t)b * HH + gr) * FW + gc) * 64 + ci]);
                } else {
                    v8h z = {};
                    *(v8h*)ldst = z;
                }
            }
            wait_asynccnt0();
        }
        __syncthreads();

        v8f acc = {};
#pragma unroll
        for (int ky = 0; ky < 3; ++ky) {
#pragma unroll
            for (int kx = 0; kx < 3; ++kx) {
                const int col = pxg * 16 + m + kx * d;
                const _Float16* abase = &in_lds[(ky * 38 + col) * 64];
                const _Float16* bbase = &wlds[(((ky * 3 + kx) * 64) + (ocg * 16 + m)) * 64];
#pragma unroll
                for (int kc = 0; kc < 2; ++kc) {
                    V16U a, bm;
                    const int kb = kc * 32 + g * 8;
                    // A 16-bit layout: elems 0..7 -> K=kb+0..7, 8..15 -> K=kb+16..23
                    a.h2[0] = *(const v8h*)(abase + kb);
                    a.h2[1] = *(const v8h*)(abase + kb + 16);
                    // B 16-bit layout: elems 0..7 -> K=kb+0..7, 8..15 -> K=kb+16..23
                    bm.h2[0] = *(const v8h*)(bbase + kb);
                    bm.h2[1] = *(const v8h*)(bbase + kb + 16);
                    acc = __builtin_amdgcn_wmma_f32_16x16x32_f16(
                        false, a.v, false, bm.v, (short)0, acc, false, false);
                }
            }
        }

        // C/D layout: VGPR v -> row M = v + 8*g, col N = lane%16.
        // Repack through LDS so the global store is one contiguous 4 KB block.
        _Float16* op = &out_lds[(pxg * 16) * 64 + ocg * 16 + m];
#pragma unroll
        for (int vv = 0; vv < 8; ++vv)
            op[(vv + 8 * g) * 64] = (_Float16)acc[vv];
        __syncthreads();

        _Float16* gout = out + (((size_t)b * HH + h) * FW + wbase) * 64;
        *(v8h*)(gout + tid * 8) = *(const v8h*)(out_lds + tid * 8);
    }
}

// ----------------------------- 1x1 conv over concat(x1,x2,x3) + residual (F)
__global__ __launch_bounds__(256) void fuse_kernel(const _Float16* __restrict__ x1,
                                                   const _Float16* __restrict__ x2,
                                                   const _Float16* __restrict__ x3,
                                                   const float* __restrict__ w4,
                                                   float* __restrict__ F) {
    __shared__ _Float16 w4lds[192 * 64];   // [co][k], 24 KB
    __shared__ _Float16 in_lds[32 * 192];  // [px][k], 12 KB

    const int tid = threadIdx.x;
    for (int t = tid; t < 192 * 64; t += 256) {
        int co = t / 192;
        int k  = t % 192;
        w4lds[t] = (_Float16)w4[(size_t)k * 64 + co];
    }

    const int lane = tid & 31, wave = tid >> 5;
    const int m = lane & 15, g = lane >> 4;
    const int pxg = wave & 1, ocg = wave >> 1;
    const int TOTAL_TILES = BB * HH * FW / 32;  // 12288

    for (int tile = blockIdx.x; tile < TOTAL_TILES; tile += gridDim.x) {
        const size_t pb = (size_t)tile * 32;
        __syncthreads();
        // stage 32 px x 192 ch: async global->LDS copies (no conversion needed)
        for (int t = tid; t < 32 * 192 / 8; t += 256) {
            int idx = t * 8;
            int lp  = idx / 192;
            int k   = idx % 192;
            int src = k >> 6;
            int ci  = k & 63;
            const _Float16* xs = (src == 0) ? x1 : (src == 1) ? x2 : x3;
            async_copy_b128(&in_lds[idx], &xs[(pb + lp) * 64 + ci]);
        }
        wait_asynccnt0();
        __syncthreads();

        v8f acc = {};
        const _Float16* abase = &in_lds[(pxg * 16 + m) * 192];
        const _Float16* bbase = &w4lds[(ocg * 16 + m) * 192];
#pragma unroll
        for (int kc = 0; kc < 6; ++kc) {
            V16U a, bm;
            const int kb = kc * 32 + g * 8;
            a.h2[0] = *(const v8h*)(abase + kb);
            a.h2[1] = *(const v8h*)(abase + kb + 16);
            bm.h2[0] = *(const v8h*)(bbase + kb);
            bm.h2[1] = *(const v8h*)(bbase + kb + 16);
            acc = __builtin_amdgcn_wmma_f32_16x16x32_f16(
                false, a.v, false, bm.v, (short)0, acc, false, false);
        }

        // residual add, in-place into F (each (px,oc) owned by exactly one lane)
        float* fp = F + (pb + pxg * 16) * 64 + ocg * 16 + m;
#pragma unroll
        for (int vv = 0; vv < 8; ++vv) {
            size_t idx = (size_t)(vv + 8 * g) * 64;
            fp[idx] = acc[vv] + fp[idx];
        }
    }
}

// ------------------------------------------------- gather per-point features
__global__ void gather_kernel(const float* __restrict__ cp,
                              const float* __restrict__ F,
                              float* __restrict__ out) {
    int i = blockIdx.x * blockDim.x + threadIdx.x;
    if (i >= NPTS) return;
    int bi, u, v;
    project_pt(cp, i, bi, u, v);
    float4* d = (float4*)(out + (size_t)i * CC);
    if (u >= FC0 && u < FC0 + FW) {
        const float4* s =
            (const float4*)(F + ((((size_t)bi * HH + v) * FW) + (u - FC0)) * CC);
#pragma unroll
        for (int c = 0; c < 16; ++c) d[c] = s[c];
    } else {
        float4 z = make_float4(0.f, 0.f, 0.f, 0.f);
#pragma unroll
        for (int c = 0; c < 16; ++c) d[c] = z;
    }
}

// --------------------------------------------------------------------- entry
extern "C" void kernel_launch(void* const* d_in, const int* in_sizes, int n_in,
                              void* d_out, int out_size, void* d_ws, size_t ws_size,
                              hipStream_t stream) {
    const float* cp = (const float*)d_in[0];  // (N,4) colored_points
    const float* pf = (const float*)d_in[1];  // (N,64) point_features
    const float* w1 = (const float*)d_in[2];  // (3,3,64,64)
    const float* w2 = (const float*)d_in[3];
    const float* w3 = (const float*)d_in[4];
    const float* w4 = (const float*)d_in[5];  // (1,1,192,64)
    float* out = (float*)d_out;

    const size_t FSZ = (size_t)BB * HH * FW * CC;  // front-region element count
    float*     F  = (float*)d_ws;                  // fp32, 100.7 MB
    _Float16*  x1 = (_Float16*)(F + FSZ);          // f16, 50.3 MB each
    _Float16*  x2 = x1 + FSZ;
    _Float16*  x3 = x2 + FSZ;

    zero_kernel<<<2048, 256, 0, stream>>>((float4*)F, (long)(FSZ / 4));
    scatter_kernel<<<(NPTS + 255) / 256, 256, 0, stream>>>(cp, pf, F);

    dim3 cgrid(FW / 32, HH / 12, BB);  // 32 px-tiles x 4 row-strips x 8 batches
    conv3x3_kernel<true ><<<cgrid, 256, 0, stream>>>(F,  w1, x1, 1);
    conv3x3_kernel<false><<<cgrid, 256, 0, stream>>>(x1, w2, x2, 2);
    conv3x3_kernel<false><<<cgrid, 256, 0, stream>>>(x2, w3, x3, 3);

    fuse_kernel<<<1024, 256, 0, stream>>>(x1, x2, x3, w4, F);
    gather_kernel<<<(NPTS + 255) / 256, 256, 0, stream>>>(cp, F, out);
}
